// DeltaNetDiT_86217173499909
// MI455X (gfx1250) — matrix operand order repaired and checked by reading, not compiled, not run on hardware
//
#include <hip/hip_runtime.h>
#include <hip/hip_bf16.h>

#define DD     768
#define NTOK   256
#define BATCH  4
#define TOK    1024   // BATCH * NTOK
#define HEADS  12
#define HDIM   64
#define EXP    8
#define CAP    320    // int(1.25 * 2 * 1024 / 8)
#define FFN    3072

typedef __bf16 v16bf  __attribute__((ext_vector_type(16)));
typedef __bf16 bf16x8 __attribute__((ext_vector_type(8)));
typedef __bf16 bf16x4 __attribute__((ext_vector_type(4)));
typedef float  v8f    __attribute__((ext_vector_type(8)));
typedef float  f32x4  __attribute__((ext_vector_type(4)));

enum { ACT_NONE = 0, ACT_GELU = 1, ACT_SILU = 2 };

// ---------------------------------------------------------------------------
// Tiled bf16 WMMA GEMM: C[M,N] = act(A[M,K] @ B[K,N] + bias), batched over z.
//
// Block tile 128x64, 8 wave32s, each wave owns a 32x32 tile -> 4 WMMAs/k-step.
// Software-pipelined: double-buffered LDS (ping/pong), next K-tile is fetched
// into registers BEFORE the current tile's fragment loads + WMMAs, then
// converted fp32->bf16 and stored; one barrier per k-step. Every WMMA fragment
// is two ds_load_b128 (A row-major, B transposed BsT[n][k] in LDS).
// Requires K % 32 == 0; M handled by clamping; NG=true guards N.
// ---------------------------------------------------------------------------
template <int ACT, bool NG>
__global__ __launch_bounds__(256) void gemm_bf16_wmma(
    const float* __restrict__ A, const float* __restrict__ B,
    const float* __restrict__ bias, float* __restrict__ C,
    int M, int N, int K,
    long long sA, long long sB, long long sBias, long long sC)
{
    const int z = blockIdx.z;
    A += (long long)z * sA;
    B += (long long)z * sB;
    C += (long long)z * sC;
    const float* biasz = bias ? bias + (long long)z * sBias : nullptr;

    // 40-element pitch = 80 bytes (16B-aligned rows, bank-spread); double buf
    __shared__ __attribute__((aligned(16))) __bf16 As[2][128][40];
    __shared__ __attribute__((aligned(16))) __bf16 BsT[2][64][40];

    const int tid  = threadIdx.x;
    const int lane = tid & 31;
    const int wave = tid >> 5;
    const int wm   = wave & 3;      // wave M tile: 32*wm
    const int wn   = wave >> 2;     // wave N tile: 32*wn
    const int m0   = blockIdx.y * 128;
    const int n0   = blockIdx.x * 64;

    // B staging coordinates: thread covers column n, K-chunk of 8
    const int bn_st = tid & 63;
    const int bk_st = (tid >> 6) << 3;              // 0,8,16,24
    const int bcol  = n0 + bn_st;
    const int bcolc = (bcol < N) ? bcol : (N - 1);  // clamp (branchless)
    const float bmask = (bcol < N) ? 1.f : 0.f;

    v8f acc[2][2];
    acc[0][0] = v8f{}; acc[0][1] = v8f{}; acc[1][0] = v8f{}; acc[1][1] = v8f{};

    f32x4 ra[4];    // in-flight A tile (16 floats/thread)
    float rb[8];    // in-flight B tile (8 floats/thread)

    auto fetch_tile = [&](int kk2) {
        #pragma unroll
        for (int i = 0; i < 4; ++i) {
            int qi = tid + i * 256;          // 1024 quads: 128 rows x 8 quads
            int r  = qi >> 3;
            int qc = (qi & 7) << 2;
            int row = m0 + r;
            row = (row < M) ? row : (M - 1); // clamp (stores guarded later)
            ra[i] = *(const f32x4*)&A[(long long)row * K + kk2 + qc];
        }
        const float* bp = B + (long long)(kk2 + bk_st) * N + (NG ? bcolc : bcol);
        #pragma unroll
        for (int j = 0; j < 8; ++j) rb[j] = bp[(long long)j * N];
        // speculative prefetch one more tile ahead (global_prefetch_b8)
        if (kk2 + 32 < K) {
            int prow = m0 + (tid >> 1);
            if (prow < M)
                __builtin_prefetch(&A[(long long)prow * K + kk2 + 32], 0, 1);
        }
    };

    auto stage_tile = [&](int buf) {
        #pragma unroll
        for (int i = 0; i < 4; ++i) {
            int qi = tid + i * 256;
            int r  = qi >> 3;
            int qc = (qi & 7) << 2;
            bf16x4 p;
            p[0] = (__bf16)ra[i][0]; p[1] = (__bf16)ra[i][1];
            p[2] = (__bf16)ra[i][2]; p[3] = (__bf16)ra[i][3];
            *(bf16x4*)&As[buf][r][qc] = p;
        }
        bf16x8 pb;
        #pragma unroll
        for (int j = 0; j < 8; ++j)
            pb[j] = (__bf16)(NG ? rb[j] * bmask : rb[j]);
        *(bf16x8*)&BsT[buf][bn_st][bk_st] = pb;
    };

    // prologue: tile 0 into buffer 0
    fetch_tile(0);
    stage_tile(0);
    __syncthreads();

    const int nsteps = K >> 5;
    for (int s = 0; s < nsteps; ++s) {
        const int cur = s & 1;
        const bool has_next = (s + 1) < nsteps;
        if (has_next) fetch_tile((s + 1) << 5);   // global loads in flight

        // ---- fragments from current buffer: each two ds_load_b128 ----
        // A 16x32: lanes 0-15 -> K[0..7]+[16..23], lanes 16-31 -> K[8..15]+[24..31]
        const int ar  = lane & 15;
        const int akb = (lane >> 4) << 3;   // 0 or 8
        v16bf afrag[2];
        #pragma unroll
        for (int f = 0; f < 2; ++f) {
            int rr = wm * 32 + f * 16 + ar;
            bf16x8 lo = *(const bf16x8*)&As[cur][rr][akb];
            bf16x8 hi = *(const bf16x8*)&As[cur][rr][16 + akb];
            afrag[f] = __builtin_shufflevector(lo, hi,
                0, 1, 2, 3, 4, 5, 6, 7, 8, 9, 10, 11, 12, 13, 14, 15);
        }
        // B 32x16: lanes 0-15 -> K[0..7]+[16..23], lanes 16-31 -> K[8..15]+[24..31]
        const int bn  = lane & 15;
        const int bkb = (lane >> 4) << 3;   // 0 or 8
        v16bf bfrag[2];
        #pragma unroll
        for (int g = 0; g < 2; ++g) {
            int nn = wn * 32 + g * 16 + bn;
            bf16x8 lo = *(const bf16x8*)&BsT[cur][nn][bkb];
            bf16x8 hi = *(const bf16x8*)&BsT[cur][nn][16 + bkb];
            bfrag[g] = __builtin_shufflevector(lo, hi,
                0, 1, 2, 3, 4, 5, 6, 7, 8, 9, 10, 11, 12, 13, 14, 15);
        }
        #pragma unroll
        for (int f = 0; f < 2; ++f)
            #pragma unroll
            for (int g = 0; g < 2; ++g)
                acc[f][g] = __builtin_amdgcn_wmma_f32_16x16x32_bf16(
                    false, afrag[f], false, bfrag[g], (short)0, acc[f][g],
                    false, false);

        if (has_next) {
            stage_tile(1 - cur);   // convert + store next tile (disjoint buffer)
            __syncthreads();       // single barrier per k-step
        }
    }

    // C layout: lanes 0-15 -> M = r, lanes 16-31 -> M = r + 8; N = lane & 15
    const int cn  = lane & 15;
    const int cmo = (lane >> 4) << 3;
    #pragma unroll
    for (int f = 0; f < 2; ++f) {
        #pragma unroll
        for (int r = 0; r < 8; ++r) {
            int row = m0 + wm * 32 + f * 16 + cmo + r;
            #pragma unroll
            for (int g = 0; g < 2; ++g) {
                int col = n0 + wn * 32 + g * 16 + cn;
                if (row < M && col < N) {
                    float v = acc[f][g][r];
                    if (biasz) v += biasz[col];
                    if (ACT == ACT_GELU) {
                        float x = v;
                        v = 0.5f * x * (1.f + tanhf(0.7978845608f *
                                                    (x + 0.044715f * x * x * x)));
                    } else if (ACT == ACT_SILU) {
                        v = v / (1.f + expf(-v));
                    }
                    C[(long long)row * N + col] = v;
                }
            }
        }
    }
}

// ---------------------------------------------------------------------------
// Patch embed + positional embed: h[b,n,d]
// ---------------------------------------------------------------------------
__global__ void patch_embed_kernel(const float* __restrict__ x,
                                   const float* __restrict__ Wpe,
                                   const float* __restrict__ bpe,
                                   const float* __restrict__ pos,
                                   float* __restrict__ h)
{
    long long id = blockIdx.x * 256LL + threadIdx.x;
    if (id >= (long long)TOK * DD) return;
    int d  = (int)(id % DD);
    int tn = (int)(id / DD);
    int b = tn >> 8, n = tn & 255;
    int pr = n >> 4, pc = n & 15;
    float acc = bpe[d];
    #pragma unroll
    for (int c = 0; c < 4; ++c)
        #pragma unroll
        for (int i = 0; i < 2; ++i)
            #pragma unroll
            for (int j = 0; j < 2; ++j)
                acc += x[(((long long)b * 4 + c) * 32 + (pr * 2 + i)) * 32 + (pc * 2 + j)]
                     * Wpe[((d * 4 + c) * 2 + i) * 2 + j];
    h[id] = acc + pos[n * DD + d];
}

// ---------------------------------------------------------------------------
// Sinusoidal time embedding + 2-layer MLP + null emb; stores cond = silu(temb)
// ---------------------------------------------------------------------------
__global__ __launch_bounds__(256) void time_embed_kernel(
    const float* __restrict__ t_in, const float* __restrict__ Wt1,
    const float* __restrict__ bt1, const float* __restrict__ Wt2,
    const float* __restrict__ bt2, const float* __restrict__ null_emb,
    float* __restrict__ cond)
{
    int b = blockIdx.x;
    __shared__ float te[DD];
    __shared__ float hb[DD];
    float tv = t_in[b];
    const int half = DD / 2;
    for (int i = threadIdx.x; i < half; i += 256) {
        float fr = expf(-logf(10000.f) * (float)i / (float)half);
        float a = tv * fr;
        te[i] = sinf(a);
        te[half + i] = cosf(a);
    }
    __syncthreads();
    for (int j = threadIdx.x; j < DD; j += 256) {
        float acc = bt1[j];
        for (int i = 0; i < DD; ++i) acc += te[i] * Wt1[i * DD + j];
        hb[j] = acc / (1.f + expf(-acc));
    }
    __syncthreads();
    for (int j = threadIdx.x; j < DD; j += 256) {
        float acc = bt2[j];
        for (int i = 0; i < DD; ++i) acc += hb[i] * Wt2[i * DD + j];
        acc += null_emb[j];
        cond[b * DD + j] = acc / (1.f + expf(-acc));
    }
}

// ---------------------------------------------------------------------------
// mods = cond @ W + bias   ([BATCH, Ncols])
// ---------------------------------------------------------------------------
__global__ void matvec_cond_kernel(const float* __restrict__ cond,
                                   const float* __restrict__ W,
                                   const float* __restrict__ bias,
                                   float* __restrict__ out, int Ncols)
{
    int id = blockIdx.x * 256 + threadIdx.x;
    if (id >= BATCH * Ncols) return;
    int b = id / Ncols, j = id % Ncols;
    float acc = bias[j];
    const float* cr = cond + b * DD;
    for (int i = 0; i < DD; ++i) acc += cr[i] * W[(long long)i * Ncols + j];
    out[id] = acc;
}

// ---------------------------------------------------------------------------
// out = layernorm(x) * (1 + sc) + sh      (one block per token)
// ---------------------------------------------------------------------------
__global__ __launch_bounds__(256) void ln_mod_kernel(
    const float* __restrict__ x, const float* __restrict__ mods,
    int mstride, int sh_off, int sc_off, float* __restrict__ out)
{
    int t = blockIdx.x;
    int b = t >> 8;
    const float* xr = x + (long long)t * DD;
    __shared__ float red[256];
    float s = 0.f;
    for (int i = threadIdx.x; i < DD; i += 256) s += xr[i];
    red[threadIdx.x] = s;
    __syncthreads();
    for (int o = 128; o > 0; o >>= 1) {
        if (threadIdx.x < o) red[threadIdx.x] += red[threadIdx.x + o];
        __syncthreads();
    }
    float mean = red[0] / (float)DD;
    __syncthreads();
    float v = 0.f;
    for (int i = threadIdx.x; i < DD; i += 256) {
        float d = xr[i] - mean;
        v += d * d;
    }
    red[threadIdx.x] = v;
    __syncthreads();
    for (int o = 128; o > 0; o >>= 1) {
        if (threadIdx.x < o) red[threadIdx.x] += red[threadIdx.x + o];
        __syncthreads();
    }
    float rstd = rsqrtf(red[0] / (float)DD + 1e-6f);
    const float* mrow = mods + (long long)b * mstride;
    for (int i = threadIdx.x; i < DD; i += 256) {
        float nv = (xr[i] - mean) * rstd;
        out[(long long)t * DD + i] = nv * (1.f + mrow[sc_off + i]) + mrow[sh_off + i];
    }
}

// h += g * z   (g broadcast per batch from mods)
__global__ void residual_kernel(float* __restrict__ h, const float* __restrict__ zv,
                                const float* __restrict__ mods, int mstride, int g_off)
{
    long long id = blockIdx.x * 256LL + threadIdx.x;
    if (id >= (long long)TOK * DD) return;
    int t = (int)(id / DD), i = (int)(id % DD), b = t >> 8;
    h[id] += mods[(long long)b * mstride + g_off + i] * zv[id];
}

__global__ void sigmoid_kernel(float* __restrict__ x, long long n)
{
    long long id = blockIdx.x * 256LL + threadIdx.x;
    if (id < n) x[id] = 1.f / (1.f + expf(-x[id]));
}

// silu + per-head L2 normalize for q and k (one block of 64 threads per token)
__global__ __launch_bounds__(64) void qk_post_kernel(float* __restrict__ q,
                                                     float* __restrict__ k)
{
    int t = blockIdx.x;
    int d = threadIdx.x;
    __shared__ float red[64];
    for (int h = 0; h < HEADS; ++h) {
        long long idx = (long long)t * DD + h * HDIM + d;
        float qv = q[idx]; qv = qv / (1.f + expf(-qv));
        float kv = k[idx]; kv = kv / (1.f + expf(-kv));
        red[d] = qv * qv; __syncthreads();
        for (int o = 32; o > 0; o >>= 1) {
            if (d < o) red[d] += red[d + o];
            __syncthreads();
        }
        float qn = sqrtf(red[0]) + 1e-6f;
        __syncthreads();
        red[d] = kv * kv; __syncthreads();
        for (int o = 32; o > 0; o >>= 1) {
            if (d < o) red[d] += red[d + o];
            __syncthreads();
        }
        float kn = sqrtf(red[0]) + 1e-6f;
        __syncthreads();
        q[idx] = qv / qn;
        k[idx] = kv / kn;
    }
}

// ---------------------------------------------------------------------------
// DeltaNet delta-rule recurrence: one block per (batch, head); S in LDS.
// S += beta * outer(k, v - k@S); o_t = q_t @ S
// ---------------------------------------------------------------------------
__global__ __launch_bounds__(256) void delta_attn_kernel(
    const float* __restrict__ q, const float* __restrict__ k,
    const float* __restrict__ v, const float* __restrict__ beta,
    float* __restrict__ o)
{
    int bh = blockIdx.x;
    int b = bh / HEADS, hh = bh % HEADS;
    __shared__ float S[HDIM][HDIM + 1];
    __shared__ float kv[HDIM], qv[HDIM], vv[HDIM], uv[HDIM];
    int tid = threadIdx.x;
    for (int i = tid; i < HDIM * HDIM; i += 256) S[i / HDIM][i % HDIM] = 0.f;
    __syncthreads();
    int jcol = tid & 63;
    int ib = (tid >> 6) * 16;
    for (int t = 0; t < NTOK; ++t) {
        long long base = ((long long)(b * NTOK + t)) * DD + hh * HDIM;
        if (tid < HDIM) {
            kv[tid] = k[base + tid];
            qv[tid] = q[base + tid];
            vv[tid] = v[base + tid];
        }
        __syncthreads();
        float bt = beta[(long long)(b * NTOK + t) * HEADS + hh];
        if (tid < HDIM) {
            float ks = 0.f;
            #pragma unroll 8
            for (int i = 0; i < HDIM; ++i) ks += kv[i] * S[i][tid];
            uv[tid] = bt * (vv[tid] - ks);
        }
        __syncthreads();
        float uj = uv[jcol];
        #pragma unroll
        for (int ii = 0; ii < 16; ++ii) S[ib + ii][jcol] += kv[ib + ii] * uj;
        __syncthreads();
        if (tid < HDIM) {
            float ov = 0.f;
            #pragma unroll 8
            for (int i = 0; i < HDIM; ++i) ov += qv[i] * S[i][tid];
            o[base + tid] = ov;
        }
        __syncthreads();
    }
}

// ---------------------------------------------------------------------------
// MoE: init / gate+top2+capacity / gather / combine / aux loss
// ---------------------------------------------------------------------------
__global__ void moe_init_kernel(int* counts, int* assign, float* prob_sum, int* slot_tok)
{
    int id = blockIdx.x * 256 + threadIdx.x;
    if (id < EXP) { counts[id] = 0; assign[id] = 0; prob_sum[id] = 0.f; }
    if (id < EXP * CAP) slot_tok[id] = -1;
}

__global__ __launch_bounds__(256) void moe_gate_kernel(
    const float* __restrict__ x2, const float* __restrict__ Wg,
    int* counts, int* assign, float* prob_sum,
    int* slot_tok, int* tok_e, int* tok_c, float* tok_w)
{
    int t = blockIdx.x * 256 + threadIdx.x;
    if (t >= TOK) return;
    float lg[EXP];
    #pragma unroll
    for (int e = 0; e < EXP; ++e) lg[e] = 0.f;
    const float* xr = x2 + (long long)t * DD;
    for (int i = 0; i < DD; ++i) {
        float xv = xr[i];
        #pragma unroll
        for (int e = 0; e < EXP; ++e) lg[e] += xv * Wg[i * EXP + e];
    }
    float mx = lg[0];
    #pragma unroll
    for (int e = 1; e < EXP; ++e) mx = fmaxf(mx, lg[e]);
    float p[EXP];
    float sum = 0.f;
    #pragma unroll
    for (int e = 0; e < EXP; ++e) { p[e] = expf(lg[e] - mx); sum += p[e]; }
    #pragma unroll
    for (int e = 0; e < EXP; ++e) { p[e] /= sum; atomicAdd(&prob_sum[e], p[e]); }
    int i0 = 0;
    for (int e = 1; e < EXP; ++e) if (p[e] > p[i0]) i0 = e;
    int i1 = (i0 == 0) ? 1 : 0;
    for (int e = 0; e < EXP; ++e) if (e != i0 && p[e] > p[i1]) i1 = e;
    float w0 = p[i0], w1 = p[i1], ws = w0 + w1;
    w0 /= ws; w1 /= ws;
    int es[2] = { i0, i1 };
    float wv[2] = { w0, w1 };
    for (int j = 0; j < 2; ++j) {
        int e = es[j];
        atomicAdd(&assign[e], 1);
        int pos = atomicAdd(&counts[e], 1);
        int c = (pos < CAP) ? pos : -1;
        if (c >= 0) slot_tok[e * CAP + c] = t;
        tok_e[t * 2 + j] = e;
        tok_c[t * 2 + j] = c;
        tok_w[t * 2 + j] = wv[j];
    }
}

__global__ void moe_gather_kernel(const float* __restrict__ x2,
                                  const int* __restrict__ slot_tok,
                                  float* __restrict__ xe)
{
    long long id = blockIdx.x * 256LL + threadIdx.x;
    if (id >= (long long)EXP * CAP * DD) return;
    int d  = (int)(id % DD);
    int ec = (int)(id / DD);
    int tok = slot_tok[ec];
    xe[id] = (tok >= 0) ? x2[(long long)tok * DD + d] : 0.f;
}

__global__ void moe_combine_kernel(float* __restrict__ h, const float* __restrict__ ye,
                                   const int* __restrict__ tok_e, const int* __restrict__ tok_c,
                                   const float* __restrict__ tok_w,
                                   const float* __restrict__ mods, int mstride, int g_off)
{
    long long id = blockIdx.x * 256LL + threadIdx.x;
    if (id >= (long long)TOK * DD) return;
    int t = (int)(id / DD), d = (int)(id % DD), b = t >> 8;
    float y = 0.f;
    #pragma unroll
    for (int j = 0; j < 2; ++j) {
        int c = tok_c[t * 2 + j];
        if (c >= 0) {
            int e = tok_e[t * 2 + j];
            y += tok_w[t * 2 + j] * ye[((long long)e * CAP + c) * DD + d];
        }
    }
    h[id] += mods[(long long)b * mstride + g_off + d] * y;
}

__global__ void moe_aux_kernel(const int* __restrict__ assign,
                               const float* __restrict__ prob_sum,
                               float* __restrict__ aux_out)
{
    if (threadIdx.x == 0 && blockIdx.x == 0) {
        float lb = 0.f;
        for (int e = 0; e < EXP; ++e) {
            float fa = ((float)assign[e] / (float)TOK) * 0.5f;   // mean(assign)/K
            float fp = prob_sum[e] / (float)TOK;                 // mean(probs)
            lb += fa * fp;
        }
        aux_out[0] += 0.01f * (float)EXP * lb;
    }
}

__global__ void zero_aux_kernel(float* __restrict__ out)
{
    if (threadIdx.x == 0 && blockIdx.x == 0) out[32768] = 0.f;
}

// unpatchify: tmp[tok, P*P*OUT_CH] -> out[B, OUT_CH, 32, 32]
__global__ void unpatch_kernel(const float* __restrict__ tmp, float* __restrict__ out)
{
    int id = blockIdx.x * 256 + threadIdx.x;
    if (id >= BATCH * 8 * 32 * 32) return;
    int xw = id & 31;
    int yh = (id >> 5) & 31;
    int o  = (id >> 10) & 7;
    int b  = id >> 13;
    int pr = yh >> 1, i = yh & 1, pc = xw >> 1, j = xw & 1;
    int tok = b * NTOK + pr * 16 + pc;
    out[id] = tmp[(long long)tok * 32 + (i * 16 + j * 8 + o)];
}

// ---------------------------------------------------------------------------
// Host launch
// ---------------------------------------------------------------------------
extern "C" void kernel_launch(void* const* d_in, const int* in_sizes, int n_in,
                              void* d_out, int out_size, void* d_ws, size_t ws_size,
                              hipStream_t stream)
{
    (void)in_sizes; (void)n_in; (void)out_size; (void)ws_size;

    const float* x        = (const float*)d_in[0];
    const float* t_in     = (const float*)d_in[1];
    const float* Wpe      = (const float*)d_in[2];
    const float* bpe      = (const float*)d_in[3];
    const float* pos      = (const float*)d_in[4];
    const float* Wt1      = (const float*)d_in[5];
    const float* bt1      = (const float*)d_in[6];
    const float* Wt2      = (const float*)d_in[7];
    const float* bt2      = (const float*)d_in[8];
    const float* null_emb = (const float*)d_in[9];
    const float* Wada     = (const float*)d_in[10];
    const float* bada     = (const float*)d_in[11];
    const float* Wq       = (const float*)d_in[12];
    const float* Wk       = (const float*)d_in[13];
    const float* Wv       = (const float*)d_in[14];
    const float* Wb       = (const float*)d_in[15];
    const float* Wo       = (const float*)d_in[16];
    const float* Wg       = (const float*)d_in[17];
    const float* W1       = (const float*)d_in[18];
    const float* b1       = (const float*)d_in[19];
    const float* W2       = (const float*)d_in[20];
    const float* b2       = (const float*)d_in[21];
    const float* Wada_f   = (const float*)d_in[22];
    const float* bada_f   = (const float*)d_in[23];
    const float* Wout     = (const float*)d_in[24];
    const float* bout     = (const float*)d_in[25];

    float* out = (float*)d_out;

    // Workspace carve-out (floats, then ints)
    float* w = (float*)d_ws;
    size_t off = 0;
    float* hbuf  = w + off; off += (size_t)TOK * DD;        // h
    float* hn    = w + off; off += (size_t)TOK * DD;        // modulated ln
    float* cond  = w + off; off += (size_t)BATCH * DD;
    float* mods  = w + off; off += (size_t)BATCH * 6 * DD;
    float* modsf = w + off; off += (size_t)BATCH * 2 * DD;
    float* qb    = w + off; off += (size_t)TOK * DD;        // q -> reused as z/tmp
    float* kb    = w + off; off += (size_t)TOK * DD;
    float* vb    = w + off; off += (size_t)TOK * DD;
    float* betab = w + off; off += (size_t)TOK * HEADS;
    float* ob    = w + off; off += (size_t)TOK * DD;        // attention output
    float* xe    = w + off; off += (size_t)EXP * CAP * DD;  // reused as ye
    float* h1    = w + off; off += (size_t)EXP * CAP * FFN;
    float* prob_sum = w + off; off += EXP;
    float* tok_w = w + off; off += (size_t)TOK * 2;
    int* ip      = (int*)(w + off);
    int* counts  = ip;
    int* assign  = ip + EXP;
    int* slot_tok= ip + 2 * EXP;
    int* tok_e   = slot_tok + EXP * CAP;
    int* tok_c   = tok_e + TOK * 2;

    float* ye  = xe;   // W2 output overwrites xe (xe fully consumed by GEMM1)
    float* zb  = qb;   // attn proj output reuses q
    float* tmp = qb;   // final head output reuses q

    const long long elems = (long long)TOK * DD;
    const int gridE = (int)((elems + 255) / 256);

    zero_aux_kernel<<<1, 32, 0, stream>>>(out);
    patch_embed_kernel<<<gridE, 256, 0, stream>>>(x, Wpe, bpe, pos, hbuf);
    time_embed_kernel<<<BATCH, 256, 0, stream>>>(t_in, Wt1, bt1, Wt2, bt2, null_emb, cond);

    for (int l = 0; l < 2; ++l) {
        const float* Wada_l = Wada + (size_t)l * DD * 6 * DD;
        const float* bada_l = bada + (size_t)l * 6 * DD;
        const float* Wq_l   = Wq   + (size_t)l * DD * DD;
        const float* Wk_l   = Wk   + (size_t)l * DD * DD;
        const float* Wv_l   = Wv   + (size_t)l * DD * DD;
        const float* Wb_l   = Wb   + (size_t)l * DD * HEADS;
        const float* Wo_l   = Wo   + (size_t)l * DD * DD;
        const float* Wg_l   = Wg   + (size_t)l * DD * EXP;
        const float* W1_l   = W1   + (size_t)l * EXP * DD * FFN;
        const float* b1_l   = b1   + (size_t)l * EXP * FFN;
        const float* W2_l   = W2   + (size_t)l * EXP * FFN * DD;
        const float* b2_l   = b2   + (size_t)l * EXP * DD;

        matvec_cond_kernel<<<(BATCH * 6 * DD + 255) / 256, 256, 0, stream>>>(
            cond, Wada_l, bada_l, mods, 6 * DD);

        // attention branch
        ln_mod_kernel<<<TOK, 256, 0, stream>>>(hbuf, mods, 6 * DD, 0, DD, hn);
        dim3 gQKV((DD + 63) / 64, (TOK + 127) / 128, 1);
        gemm_bf16_wmma<ACT_NONE, false><<<gQKV, 256, 0, stream>>>(hn, Wq_l, nullptr, qb,
            TOK, DD, DD, 0, 0, 0, 0);
        gemm_bf16_wmma<ACT_NONE, false><<<gQKV, 256, 0, stream>>>(hn, Wk_l, nullptr, kb,
            TOK, DD, DD, 0, 0, 0, 0);
        gemm_bf16_wmma<ACT_NONE, false><<<gQKV, 256, 0, stream>>>(hn, Wv_l, nullptr, vb,
            TOK, DD, DD, 0, 0, 0, 0);
        dim3 gB(1, (TOK + 127) / 128, 1);
        gemm_bf16_wmma<ACT_NONE, true><<<gB, 256, 0, stream>>>(hn, Wb_l, nullptr, betab,
            TOK, HEADS, DD, 0, 0, 0, 0);
        sigmoid_kernel<<<(TOK * HEADS + 255) / 256, 256, 0, stream>>>(betab,
            (long long)TOK * HEADS);
        qk_post_kernel<<<TOK, 64, 0, stream>>>(qb, kb);
        delta_attn_kernel<<<BATCH * HEADS, 256, 0, stream>>>(qb, kb, vb, betab, ob);
        gemm_bf16_wmma<ACT_NONE, false><<<gQKV, 256, 0, stream>>>(ob, Wo_l, nullptr, zb,
            TOK, DD, DD, 0, 0, 0, 0);
        residual_kernel<<<gridE, 256, 0, stream>>>(hbuf, zb, mods, 6 * DD, 2 * DD);

        // MoE branch
        ln_mod_kernel<<<TOK, 256, 0, stream>>>(hbuf, mods, 6 * DD, 3 * DD, 4 * DD, hn);
        moe_init_kernel<<<(EXP * CAP + 255) / 256, 256, 0, stream>>>(
            counts, assign, prob_sum, slot_tok);
        moe_gate_kernel<<<TOK / 256, 256, 0, stream>>>(hn, Wg_l, counts, assign,
            prob_sum, slot_tok, tok_e, tok_c, tok_w);
        moe_gather_kernel<<<(int)(((long long)EXP * CAP * DD + 255) / 256), 256, 0, stream>>>(
            hn, slot_tok, xe);
        dim3 g1((FFN + 63) / 64, (CAP + 127) / 128, EXP);
        gemm_bf16_wmma<ACT_GELU, false><<<g1, 256, 0, stream>>>(xe, W1_l, b1_l, h1,
            CAP, FFN, DD,
            (long long)CAP * DD, (long long)DD * FFN, FFN, (long long)CAP * FFN);
        dim3 g2((DD + 63) / 64, (CAP + 127) / 128, EXP);
        gemm_bf16_wmma<ACT_NONE, false><<<g2, 256, 0, stream>>>(h1, W2_l, b2_l, ye,
            CAP, DD, FFN,
            (long long)CAP * FFN, (long long)FFN * DD, DD, (long long)CAP * DD);
        moe_combine_kernel<<<gridE, 256, 0, stream>>>(hbuf, ye, tok_e, tok_c, tok_w,
            mods, 6 * DD, 5 * DD);
        moe_aux_kernel<<<1, 32, 0, stream>>>(assign, prob_sum, out + 32768);
    }

    // final modulation + head + unpatchify
    matvec_cond_kernel<<<(BATCH * 2 * DD + 255) / 256, 256, 0, stream>>>(
        cond, Wada_f, bada_f, modsf, 2 * DD);
    ln_mod_kernel<<<TOK, 256, 0, stream>>>(hbuf, modsf, 2 * DD, 0, DD, hn);
    dim3 gO(1, (TOK + 127) / 128, 1);
    gemm_bf16_wmma<ACT_NONE, true><<<gO, 256, 0, stream>>>(hn, Wout, bout, tmp,
        TOK, 32, DD, 0, 0, 0, 0);
    unpatch_kernel<<<(BATCH * 8 * 32 * 32 + 255) / 256, 256, 0, stream>>>(tmp, out);
}
